// GraphSAGENetSkip_13099650253556
// MI455X (gfx1250) — matrix operand order repaired and checked by reading, not compiled, not run on hardware
//
#include <hip/hip_runtime.h>
#include <hip/hip_bf16.h>

#define N_NODES 50000
#define N_EDGES 800000
#define NF      128
#define OUT_CH  3
#define EPS     1e-5f
#define SLOPE   0.02f
#define LDS_F   140   // f32 staging row stride: 560B = 35*16B aligned, 12-bank skew
#define LDB     136   // bf16 tile row stride: 272B = 17*16B aligned, 4-bank skew

typedef __attribute__((ext_vector_type(16))) __bf16 v16bf;
typedef __attribute__((ext_vector_type(8)))  float  v8f;

// -inf (empty segment) -> 0, matching jnp.where(isfinite(agg), agg, 0)
__device__ __forceinline__ float fixinf(float x) {
    return (x < -1.0e30f) ? 0.0f : x;
}

// Async memory->LDS b128 transfer (gfx1250 VFLAT async op, tracked by ASYNCcnt).
__device__ __forceinline__ void async_ld_b128(unsigned int lds_byte_off,
                                              unsigned long long gaddr) {
    asm volatile("global_load_async_to_lds_b128 %0, %1, off"
                 :: "v"(lds_byte_off), "v"(gaddr) : "memory");
}
__device__ __forceinline__ void wait_asynccnt0() {
    asm volatile("s_wait_asynccnt 0x0" ::: "memory");
}

// ---------------------------------------------------------------------------
// A-fragment (16x32 bf16, MxK) from the packed bf16 LDS tile, K window at kk.
// ISA 7.12.2: lanes 0-15 rows M=lane, local K 0-7 in v0..3 / 16-23 in v4..7;
// lanes 16-31 same rows, local K +8. Two contiguous 16B runs per lane
// -> exactly 2x ds_load_b128, no VALU.
// ---------------------------------------------------------------------------
__device__ __forceinline__ v16bf load_a_frag(const __bf16* sT, int lane, int kk) {
    const int m    = lane & 15;
    const int koff = (lane >> 4) * 8;
    const __bf16* p = sT + m * LDB + kk + koff;
    v16bf a;
#pragma unroll
    for (int i = 0; i < 8; ++i) a[i] = p[i];
#pragma unroll
    for (int i = 0; i < 8; ++i) a[8 + i] = p[16 + i];
    return a;
}

// ---------------------------------------------------------------------------
// Weight pre-pack: fp32 W[128][128] -> bf16 fragments in B-matrix lane order.
// Fragment index = (kkIdx*8 + colTile)*32 + lane; each holds 16 bf16 (32B)
// so the GEMM reads one contiguous 32B chunk per lane (coalesced 1KB/wave).
// b[s] = W[kk + (lane>=16)*8 + s + (s>=8)*8][colTile*16 + (lane&15)]
// ---------------------------------------------------------------------------
__global__ __launch_bounds__(256) void pack_weights(const float* __restrict__ W,
                                                    __bf16* __restrict__ P) {
    const int tid  = blockIdx.x * 256 + threadIdx.x;   // 0..1023 (grid = 4)
    const int kkI  = tid >> 8;
    const int rem  = tid & 255;
    const int w    = rem >> 5;
    const int lane = rem & 31;
    const int kk = kkI * 32 + (lane >> 4) * 8;
    const int n  = w * 16 + (lane & 15);
    __bf16* dst = P + (size_t)tid * 16;
#pragma unroll
    for (int i = 0; i < 16; ++i)
        dst[i] = (__bf16)W[(size_t)(kk + i + (i >= 8 ? 8 : 0)) * NF + n];
}

// ---------------------------------------------------------------------------
// Out[16 rows x 128 cols] = fix(agg) @ Wl + h @ Wr + bias, via bf16 WMMA.
// Phase 1: async-to-LDS DMA of the f32 tiles. Phase 2: single fix+convert
// sweep into packed bf16 tiles. Phase 3: pure ds_load_b128 + WMMA K-loop.
// ---------------------------------------------------------------------------
__global__ __launch_bounds__(256) void gemm_sage(const float* __restrict__ Agg,
                                                 const float* __restrict__ H,
                                                 const __bf16* __restrict__ Pl,
                                                 const __bf16* __restrict__ Pr,
                                                 const float* __restrict__ bias,
                                                 float* __restrict__ Out) {
    __shared__ float  fA[16 * LDS_F];
    __shared__ float  fH[16 * LDS_F];
    __shared__ __bf16 sA[16 * LDB];
    __shared__ __bf16 sH[16 * LDB];

    const int row0 = blockIdx.x * 16;
    const int t    = threadIdx.x;

    // Phase 1: stage 16x128 agg + h tiles (4 async b128 transfers per thread).
#pragma unroll
    for (int r = 0; r < 2; ++r) {
        const int linear = t + r * 256;        // float4 index, 0..511
        const int row    = linear >> 5;        // 32 float4 per row
        const int c4     = (linear & 31) * 4;
        const unsigned int la = (unsigned int)(uintptr_t)&fA[row * LDS_F + c4];
        const unsigned int lh = (unsigned int)(uintptr_t)&fH[row * LDS_F + c4];
        async_ld_b128(la, (unsigned long long)(uintptr_t)(Agg + (size_t)(row0 + row) * NF + c4));
        async_ld_b128(lh, (unsigned long long)(uintptr_t)(H   + (size_t)(row0 + row) * NF + c4));
    }
    wait_asynccnt0();
    __syncthreads();

    // Phase 2: one fix+convert sweep, f32 -> packed bf16 tiles.
#pragma unroll
    for (int r = 0; r < 2; ++r) {
        const int linear = t + r * 256;
        const int row    = linear >> 5;
        const int c4     = (linear & 31) * 4;
        const float4 a = *(const float4*)&fA[row * LDS_F + c4];
        const float4 h = *(const float4*)&fH[row * LDS_F + c4];
        __bf16* pa = &sA[row * LDB + c4];
        __bf16* ph = &sH[row * LDB + c4];
        pa[0] = (__bf16)fixinf(a.x); pa[1] = (__bf16)fixinf(a.y);
        pa[2] = (__bf16)fixinf(a.z); pa[3] = (__bf16)fixinf(a.w);
        ph[0] = (__bf16)h.x; ph[1] = (__bf16)h.y;
        ph[2] = (__bf16)h.z; ph[3] = (__bf16)h.w;
    }
    __syncthreads();

    const int wave = t >> 5;
    const int lane = t & 31;

    // Phase 3: K loop — pure matrix work.
    v8f acc = {};
#pragma unroll
    for (int kk = 0; kk < NF; kk += 32) {
        const size_t fidx = (((size_t)(kk >> 5) * 8 + wave) * 32 + lane) * 16;
        v16bf af = load_a_frag(sA, lane, kk);
        v16bf bf = *(const v16bf*)(Pl + fidx);
        acc = __builtin_amdgcn_wmma_f32_16x16x32_bf16(false, af, false, bf,
                                                      (short)0, acc, false, false);
        v16bf hf = load_a_frag(sH, lane, kk);
        v16bf cf = *(const v16bf*)(Pr + fidx);
        acc = __builtin_amdgcn_wmma_f32_16x16x32_bf16(false, hf, false, cf,
                                                      (short)0, acc, false, false);
    }

    // C/D layout: VGPR v holds row (v + 8*(lane>=16)), col = lane&15.
    const int n     = wave * 16 + (lane & 15);
    const float bv  = bias[n];
    const int mbase = row0 + ((lane >> 4) * 8);
#pragma unroll
    for (int v = 0; v < 8; ++v)
        Out[(size_t)(mbase + v) * NF + n] = acc[v] + bv;
}

// ---------------------------------------------------------------------------
// Scatter-max aggregation: one wave covers one edge's 128-ch row (float4/lane),
// native f32 max atomics into the L2-resident destination rows.
// ---------------------------------------------------------------------------
__global__ __launch_bounds__(256) void init_neginf(float* __restrict__ p) {
    const size_t i = (size_t)blockIdx.x * blockDim.x + threadIdx.x;
    p[i] = -INFINITY;
}

__global__ __launch_bounds__(256) void scatter_max_k(const int* __restrict__ ei,
                                                     const float* __restrict__ H,
                                                     float* __restrict__ Agg) {
    const int tid = blockIdx.x * blockDim.x + threadIdx.x;
    const int e   = tid >> 5;
    const int c   = (tid & 31) * 4;
    const int s = ei[e];
    const int d = ei[N_EDGES + e];
    const float4 v = *(const float4*)(H + (size_t)s * NF + c);
    float* p = Agg + (size_t)d * NF + c;
    __hip_atomic_fetch_max(p + 0, v.x, __ATOMIC_RELAXED, __HIP_MEMORY_SCOPE_AGENT);
    __hip_atomic_fetch_max(p + 1, v.y, __ATOMIC_RELAXED, __HIP_MEMORY_SCOPE_AGENT);
    __hip_atomic_fetch_max(p + 2, v.z, __ATOMIC_RELAXED, __HIP_MEMORY_SCOPE_AGENT);
    __hip_atomic_fetch_max(p + 3, v.w, __ATOMIC_RELAXED, __HIP_MEMORY_SCOPE_AGENT);
}

// ---------------------------------------------------------------------------
// GraphNorm: two column reductions + elementwise normalize/LeakyReLU.
// ---------------------------------------------------------------------------
__global__ __launch_bounds__(128) void gn_mean(const float* __restrict__ H,
                                               float* __restrict__ mu) {
    const int t  = threadIdx.x;
    const int r0 = blockIdx.x * 250;
    float s = 0.0f;
    for (int r = 0; r < 250; ++r)
        s += H[(size_t)(r0 + r) * NF + t];
    atomicAdd(&mu[t], s);
}

__global__ __launch_bounds__(128) void gn_var(const float* __restrict__ H,
                                              const float* __restrict__ mu,
                                              float* __restrict__ var,
                                              const float* __restrict__ ms) {
    const int t  = threadIdx.x;
    const int r0 = blockIdx.x * 250;
    const float m = ms[t] * (mu[t] * (1.0f / N_NODES));
    float s2 = 0.0f;
    for (int r = 0; r < 250; ++r) {
        const float d = H[(size_t)(r0 + r) * NF + t] - m;
        s2 += d * d;
    }
    atomicAdd(&var[t], s2);
}

__global__ __launch_bounds__(256) void gn_norm(float* __restrict__ H,
                                               const float* __restrict__ mu,
                                               const float* __restrict__ var,
                                               const float* __restrict__ gamma,
                                               const float* __restrict__ beta,
                                               const float* __restrict__ ms) {
    const size_t i = (size_t)blockIdx.x * blockDim.x + threadIdx.x;
    const int f = (int)(i & (NF - 1));
    const float m  = mu[f] * (1.0f / N_NODES);
    const float rs = rsqrtf(var[f] * (1.0f / N_NODES) + EPS);
    const float s  = H[i] - ms[f] * m;
    float v = gamma[f] * s * rs + beta[f];
    H[i] = (v >= 0.0f) ? v : SLOPE * v;
}

// ---------------------------------------------------------------------------
// Tail: out = tanh(fix(aggH)@Wl[:128] + h@Wr[:128]
//              + fix(aggS)@Wl[128:] + xs@Wr[128:] + bl) * 0.5   (N=3: scalar)
// ---------------------------------------------------------------------------
__global__ __launch_bounds__(256) void tail_k(const float* __restrict__ aggH,
                                              const float* __restrict__ Hf,
                                              const float* __restrict__ aggS,
                                              const float* __restrict__ XS,
                                              const float* __restrict__ Wl,
                                              const float* __restrict__ Wr,
                                              const float* __restrict__ bl,
                                              float* __restrict__ out) {
    const int n = blockIdx.x * blockDim.x + threadIdx.x;
    if (n >= N_NODES) return;
    float a0 = bl[0], a1 = bl[1], a2 = bl[2];
    const size_t base = (size_t)n * NF;
    for (int k = 0; k < NF; ++k) {
        const float ah = fixinf(aggH[base + k]);
        const float hv = Hf[base + k];
        const float as = fixinf(aggS[base + k]);
        const float xv = XS[base + k];
        a0 += ah * Wl[k * 3 + 0] + hv * Wr[k * 3 + 0]
            + as * Wl[(NF + k) * 3 + 0] + xv * Wr[(NF + k) * 3 + 0];
        a1 += ah * Wl[k * 3 + 1] + hv * Wr[k * 3 + 1]
            + as * Wl[(NF + k) * 3 + 1] + xv * Wr[(NF + k) * 3 + 1];
        a2 += ah * Wl[k * 3 + 2] + hv * Wr[k * 3 + 2]
            + as * Wl[(NF + k) * 3 + 2] + xv * Wr[(NF + k) * 3 + 2];
    }
    out[n * 3 + 0] = tanhf(a0) * 0.5f;
    out[n * 3 + 1] = tanhf(a1) * 0.5f;
    out[n * 3 + 2] = tanhf(a2) * 0.5f;
}

// ---------------------------------------------------------------------------
extern "C" void kernel_launch(void* const* d_in, const int* in_sizes, int n_in,
                              void* d_out, int out_size, void* d_ws, size_t ws_size,
                              hipStream_t stream) {
    (void)in_sizes; (void)n_in; (void)out_size; (void)ws_size;

    const float* x        = (const float*)d_in[0];
    const int*   ei       = (const int*)d_in[1];
    const float* Wl_head  = (const float*)d_in[2];
    const float* bl_head  = (const float*)d_in[3];
    const float* Wr_head  = (const float*)d_in[4];
    const float* g_head   = (const float*)d_in[5];
    const float* b_head   = (const float*)d_in[6];
    const float* ms_head  = (const float*)d_in[7];
    const float* Wl_skip  = (const float*)d_in[8];
    const float* bl_skip  = (const float*)d_in[9];
    const float* Wr_skip  = (const float*)d_in[10];
    const float* Wl_tail  = (const float*)d_in[11];
    const float* bl_tail  = (const float*)d_in[12];
    const float* Wr_tail  = (const float*)d_in[13];
    float* out = (float*)d_out;

    const size_t NE = (size_t)N_NODES * NF;
    const size_t WSZ = (size_t)NF * NF;            // elements per packed matrix
    float*  bufA  = (float*)d_ws;
    float*  bufB  = bufA + NE;
    float*  agg   = bufB + NE;
    float*  xsk   = agg + NE;
    __bf16* packW = (__bf16*)(xsk + NE);           // 16 matrices x 32KB (256B aligned)
    float*  stats = (float*)(packW + 16 * WSZ);    // mu[128] | var[128]
    float*  mu  = stats;
    float*  var = stats + NF;

    const int scatterBlocks = (N_EDGES * 32) / 256;   // 100000
    const int elemBlocks    = (int)(NE / 256);        // 25000
    const int gemmBlocks    = N_NODES / 16;           // 3125

    // ---- pre-pack all 128x128 weight matrices to bf16 B-fragment order ----
    for (int i = 0; i < 7; ++i) {
        pack_weights<<<4, 256, 0, stream>>>(Wl_head + (size_t)i * WSZ, packW + (size_t)(2 * i) * WSZ);
        pack_weights<<<4, 256, 0, stream>>>(Wr_head + (size_t)i * WSZ, packW + (size_t)(2 * i + 1) * WSZ);
    }
    pack_weights<<<4, 256, 0, stream>>>(Wl_skip, packW + (size_t)14 * WSZ);
    pack_weights<<<4, 256, 0, stream>>>(Wr_skip, packW + (size_t)15 * WSZ);

    // ---- head: 7 SAGEConv layers, GraphNorm+LeakyReLU on 0..5 ----
    const float* h = x;
    float* bufs[2] = { bufA, bufB };
    for (int i = 0; i < 7; ++i) {
        float* hn = bufs[i & 1];
        init_neginf<<<elemBlocks, 256, 0, stream>>>(agg);
        scatter_max_k<<<scatterBlocks, 256, 0, stream>>>(ei, h, agg);
        gemm_sage<<<gemmBlocks, 256, 0, stream>>>(agg, h,
            packW + (size_t)(2 * i) * WSZ, packW + (size_t)(2 * i + 1) * WSZ,
            bl_head + (size_t)i * NF, hn);
        if (i < 6) {
            hipMemsetAsync(stats, 0, 2 * NF * sizeof(float), stream);
            gn_mean<<<200, 128, 0, stream>>>(hn, mu);
            gn_var<<<200, 128, 0, stream>>>(hn, mu, var, ms_head + (size_t)i * NF);
            gn_norm<<<elemBlocks, 256, 0, stream>>>(hn, mu, var,
                g_head + (size_t)i * NF, b_head + (size_t)i * NF,
                ms_head + (size_t)i * NF);
        }
        h = hn;
    }
    // final h lives in bufA (layers 0,2,4,6 write bufA)

    // ---- skip branch: conv on raw x ----
    init_neginf<<<elemBlocks, 256, 0, stream>>>(agg);
    scatter_max_k<<<scatterBlocks, 256, 0, stream>>>(ei, x, agg);
    gemm_sage<<<gemmBlocks, 256, 0, stream>>>(agg, x,
        packW + (size_t)14 * WSZ, packW + (size_t)15 * WSZ, bl_skip, xsk);

    // ---- tail: split-concat trick; aggH -> agg, aggS -> bufB (free) ----
    init_neginf<<<elemBlocks, 256, 0, stream>>>(agg);
    scatter_max_k<<<scatterBlocks, 256, 0, stream>>>(ei, h, agg);
    init_neginf<<<elemBlocks, 256, 0, stream>>>(bufB);
    scatter_max_k<<<scatterBlocks, 256, 0, stream>>>(ei, xsk, bufB);
    tail_k<<<(N_NODES + 255) / 256, 256, 0, stream>>>(agg, h, bufB, xsk,
                                                      Wl_tail, Wr_tail, bl_tail, out);
}
